// CustomModel_83794811945675
// MI455X (gfx1250) — matrix operand order repaired and checked by reading, not compiled, not run on hardware
//
#include <hip/hip_runtime.h>
#include <math.h>

typedef __attribute__((ext_vector_type(16))) __bf16 v16bf;
typedef __attribute__((ext_vector_type(8)))  __bf16 v8bf;
typedef __attribute__((ext_vector_type(4)))  __bf16 v4bf;
typedef __attribute__((ext_vector_type(8)))  float  v8f;

#define VOCAB 50000
#define EMB   32
#define HID   64
#define NCLS  3
#define BSZ   512
#define TLEN  512
#define NH4   256   // 4*HID
#define MT    16    // batch rows per block / vocab rows per block
#define NTHR  128   // 4 waves: wave w owns hidden slice j in [16w, 16w+16)

#define LOG2E 1.4426950408889634f

// ---- fused LSTM gate update: 5 exp + 2 rcp (trans), rest dual-issue VALU ----
// i=sig(zi), f=sig(zf), g=tanh(zg), o=sig(zo); c=f*c+i*g; returns h=o*tanh(c)
__device__ __forceinline__ float gate_update(float zi, float zf, float zg, float zo,
                                             float& cstate) {
  const float ei = __builtin_amdgcn_exp2f(fminf(-LOG2E * zi, 30.f));        // e^-zi
  const float ef = __builtin_amdgcn_exp2f(fminf(-LOG2E * zf, 30.f));        // e^-zf
  const float eo = __builtin_amdgcn_exp2f(fminf(-LOG2E * zo, 30.f));        // e^-zo
  const float eg = __builtin_amdgcn_exp2f(fminf(2.f * LOG2E * zg, 30.f));   // e^{2zg}
  const float p  = (1.f + ei) * (eg + 1.f);
  const float q  = 1.f + ef;
  const float c  = (cstate * p + (eg - 1.f) * q) * __builtin_amdgcn_rcpf(q * p);
  cstate = c;
  const float ec = __builtin_amdgcn_exp2f(fminf(2.f * LOG2E * c, 30.f));    // e^{2c}
  return (ec - 1.f) * __builtin_amdgcn_rcpf((1.f + eo) * (ec + 1.f));
}

// ---- common fragment geometry helpers (wave32) ----
// A (16-bit, 16x32): lane L -> M=L%16, Kbase=(L>=16?16:0); elems 0-15=K+0..15
// B (16-bit, 32x16): lane L -> N=L%16, K=(L>=16?16:0)+e
// C/D (f32, 16x16):  lane L -> N=L%16, rows M=r+(L>=16?8:0)

// =====================================================================
// Kernel A: embWk[v][col] = emb[v][:] @ Wk[:][col] + bias[col]   (WMMA)
// =====================================================================
__global__ __launch_bounds__(NTHR)
void emb_proj(const float* __restrict__ emb,   // [VOCAB, EMB]
              const float* __restrict__ Wk,    // [EMB, 4H]
              const float* __restrict__ bias,  // [4H]
              float*       __restrict__ zx)    // [VOCAB, 4H]
{
  const int tid  = threadIdx.x;
  const int lane = tid & 31;
  const int wave = tid >> 5;
  const int row0 = blockIdx.x * MT;            // VOCAB = 3125 * 16 exactly

  const int colInTile = lane & 15;
  const int j     = wave * 16 + colInTile;
  const int khalf = (lane >> 4) * 16;
  const int aM    = lane & 15;
  const int aK    = (lane >> 4) * 16;
  const int mr    = (lane >> 4) * 8;

  // B fragments + bias
  v16bf bWk[4];
  float bgate[4];
#pragma unroll
  for (int g = 0; g < 4; ++g) {
    bgate[g] = bias[g * HID + j];
    const int col = g * HID + j;
#pragma unroll
    for (int e = 0; e < 16; ++e) bWk[g][e] = (__bf16)Wk[(khalf + e) * NH4 + col];
  }

  // A fragment straight from global (rows are contiguous f32 runs)
  const float* er = emb + (size_t)(row0 + aM) * EMB;
  v16bf aX;
#pragma unroll
  for (int e = 0; e < 8; ++e) {
    aX[e]     = (__bf16)er[aK + e];
    aX[8 + e] = (__bf16)er[aK + 8 + e];
  }

#pragma unroll
  for (int g = 0; g < 4; ++g) {
    v8f a = {bgate[g], bgate[g], bgate[g], bgate[g], bgate[g], bgate[g], bgate[g], bgate[g]};
    a = __builtin_amdgcn_wmma_f32_16x16x32_bf16(false, aX, false, bWk[g], (short)0, a, false, false);
#pragma unroll
    for (int r = 0; r < 8; ++r)
      zx[(size_t)(row0 + mr + r) * NH4 + g * HID + j] = a[r];
  }
}

// =====================================================================
// Kernel B: recurrence with precomputed z-init (8 WMMAs/step, 2-chains)
// =====================================================================
__global__ __launch_bounds__(NTHR)
void lstm_rec(const int*   __restrict__ tokens,
              const float* __restrict__ zx,    // [VOCAB, 4H] = emb@Wk + bias (L2-resident)
              const float* __restrict__ Wr,    // [HID, 4H]
              const float* __restrict__ Wd,    // [HID, NCLS]
              const float* __restrict__ bd,    // [NCLS]
              float*       __restrict__ out)   // [BSZ, NCLS]
{
  __shared__ int               tks[MT][TLEN];          // token slice (32 KB)
  __shared__ __align__(16) __bf16 hb[2][MT][HID + 8];  // double-buffered h tile
  __shared__ float             hf[MT][HID];
  __shared__ float             lg[MT][NCLS];

  const int tid  = threadIdx.x;
  const int lane = tid & 31;
  const int wave = tid >> 5;
  const int b0   = blockIdx.x * MT;

  for (int i = tid; i < MT * TLEN; i += NTHR) ((int*)tks)[i] = tokens[b0 * TLEN + i];
  for (int i = tid; i < MT * (HID + 8); i += NTHR) ((__bf16*)hb[0])[i] = (__bf16)0.0f;

  const int colInTile = lane & 15;
  const int j     = wave * 16 + colInTile;
  const int khalf = (lane >> 4) * 16;
  const int mr    = (lane >> 4) * 8;
  const int aM    = lane & 15;
  const int aK    = (lane >> 4) * 16;

  // loop-invariant recurrent B fragments
  v16bf bWr0[4], bWr1[4];
#pragma unroll
  for (int g = 0; g < 4; ++g) {
    const int col = g * HID + j;
#pragma unroll
    for (int e = 0; e < 16; ++e) {
      bWr0[g][e] = (__bf16)Wr[(khalf + e) * NH4 + col];
      bWr1[g][e] = (__bf16)Wr[(32 + khalf + e) * NH4 + col];
    }
  }

  float creg[8], hreg[8];
#pragma unroll
  for (int r = 0; r < 8; ++r) { creg[r] = 0.f; hreg[r] = 0.f; }

  __syncthreads();   // tks + hb[0] visible

  // prefetch z-init (C operand) for t=0: per lane 8 rows x 4 gates
  v8f zc[4];
#pragma unroll
  for (int r = 0; r < 8; ++r) {
    const int tok = tks[mr + r][0];
    const float* base = zx + (size_t)tok * NH4 + j;
#pragma unroll
    for (int g = 0; g < 4; ++g) zc[g][r] = base[g * HID];
  }

  for (int t = 0; t < TLEN; ++t) {
    const int cur = t & 1, nxt = cur ^ 1;

    // h fragments
    v8bf h0l = *(const v8bf*)&hb[cur][aM][aK];
    v8bf h0h = *(const v8bf*)&hb[cur][aM][aK + 8];
    v8bf h1l = *(const v8bf*)&hb[cur][aM][32 + aK];
    v8bf h1h = *(const v8bf*)&hb[cur][aM][32 + aK + 8];
    v16bf aH0 = __builtin_shufflevector(h0l, h0h, 0,1,2,3,4,5,6,7,8,9,10,11,12,13,14,15);
    v16bf aH1 = __builtin_shufflevector(h1l, h1h, 0,1,2,3,4,5,6,7,8,9,10,11,12,13,14,15);

    // prefetch next z-init (issued before WMMAs, consumed next iteration)
    const int tp = (t + 1 < TLEN) ? (t + 1) : (TLEN - 1);
    v8f zn[4];
#pragma unroll
    for (int r = 0; r < 8; ++r) {
      const int tok = tks[mr + r][tp];
      const float* base = zx + (size_t)tok * NH4 + j;
#pragma unroll
      for (int g = 0; g < 4; ++g) zn[g][r] = base[g * HID];
    }

    // z = zc + h@Wr : 4 independent 2-WMMA chains
    v8f acc[4];
#pragma unroll
    for (int g = 0; g < 4; ++g) {
      v8f a = zc[g];
      a = __builtin_amdgcn_wmma_f32_16x16x32_bf16(false, aH0, false, bWr0[g], (short)0, a, false, false);
      a = __builtin_amdgcn_wmma_f32_16x16x32_bf16(false, aH1, false, bWr1[g], (short)0, a, false, false);
      acc[g] = a;
    }

    // gates + cell update in registers
#pragma unroll
    for (int r = 0; r < 8; ++r) {
      const float h = gate_update(acc[0][r], acc[1][r], acc[2][r], acc[3][r], creg[r]);
      hreg[r] = h;
      hb[nxt][mr + r][j] = (__bf16)h;
    }

#pragma unroll
    for (int g = 0; g < 4; ++g) zc[g] = zn[g];

    __syncthreads();
  }

#pragma unroll
  for (int r = 0; r < 8; ++r) hf[mr + r][j] = hreg[r];
  __syncthreads();

  if (tid < MT * NCLS) {
    const int m = tid / NCLS, k = tid % NCLS;
    float acc = bd[k];
#pragma unroll
    for (int jj = 0; jj < HID; ++jj) acc += hf[m][jj] * Wd[jj * NCLS + k];
    lg[m][k] = acc;
  }
  __syncthreads();
  if (tid < MT) {
    const float l0 = lg[tid][0], l1 = lg[tid][1], l2 = lg[tid][2];
    const float mx = fmaxf(l0, fmaxf(l1, l2));
    const float e0 = __builtin_amdgcn_exp2f(LOG2E * (l0 - mx));
    const float e1 = __builtin_amdgcn_exp2f(LOG2E * (l1 - mx));
    const float e2 = __builtin_amdgcn_exp2f(LOG2E * (l2 - mx));
    const float inv = 1.f / (e0 + e1 + e2);
    float* o = out + (size_t)(b0 + tid) * NCLS;
    o[0] = e0 * inv; o[1] = e1 * inv; o[2] = e2 * inv;
  }
}

// =====================================================================
// Kernel C: fallback (no workspace): fused embed + LSTM, 12 WMMAs/step
// =====================================================================
__global__ __launch_bounds__(NTHR)
void lstm_fused(const int*   __restrict__ tokens,
                const float* __restrict__ emb,   // [VOCAB, EMB] fp32
                const float* __restrict__ Wk,
                const float* __restrict__ Wr,
                const float* __restrict__ bias,
                const float* __restrict__ Wd,
                const float* __restrict__ bd,
                float*       __restrict__ out)
{
  __shared__ int               tks[MT][TLEN];
  __shared__ __align__(16) __bf16 xs[2][MT][EMB];
  __shared__ __align__(16) __bf16 hb[2][MT][HID + 8];
  __shared__ float             hf[MT][HID];
  __shared__ float             lg[MT][NCLS];

  const int tid  = threadIdx.x;
  const int lane = tid & 31;
  const int wave = tid >> 5;
  const int b0   = blockIdx.x * MT;

  for (int i = tid; i < MT * TLEN; i += NTHR) ((int*)tks)[i] = tokens[b0 * TLEN + i];
  for (int i = tid; i < MT * (HID + 8); i += NTHR) ((__bf16*)hb[0])[i] = (__bf16)0.0f;
  {
    const int r = tid >> 3, ch = tid & 7;          // 16 rows x 8 chunks of 4 f32
    const int tok = tokens[(b0 + r) * TLEN + 0];
    const float4 f = ((const float4*)(emb + (size_t)tok * EMB))[ch];
    v4bf p = {(__bf16)f.x, (__bf16)f.y, (__bf16)f.z, (__bf16)f.w};
    *(v4bf*)&xs[0][r][ch * 4] = p;
  }

  const int colInTile = lane & 15;
  const int j     = wave * 16 + colInTile;
  const int khalf = (lane >> 4) * 16;
  float bgate[4];
#pragma unroll
  for (int g = 0; g < 4; ++g) bgate[g] = bias[g * HID + j];

  v16bf bWk[4], bWr0[4], bWr1[4];
#pragma unroll
  for (int g = 0; g < 4; ++g) {
    const int col = g * HID + j;
#pragma unroll
    for (int e = 0; e < 16; ++e) {
      bWk[g][e]  = (__bf16)Wk[(khalf + e) * NH4 + col];
      bWr0[g][e] = (__bf16)Wr[(khalf + e) * NH4 + col];
      bWr1[g][e] = (__bf16)Wr[(32 + khalf + e) * NH4 + col];
    }
  }

  const int mr = (lane >> 4) * 8;
  float creg[8], hreg[8];
#pragma unroll
  for (int r = 0; r < 8; ++r) { creg[r] = 0.f; hreg[r] = 0.f; }

  const int aM = lane & 15;
  const int aK = (lane >> 4) * 16;
  const int gr = tid >> 3, gch = tid & 7;

  __syncthreads();

  for (int t = 0; t < TLEN; ++t) {
    const int cur = t & 1, nxt = cur ^ 1;

    v8bf xlo = *(const v8bf*)&xs[cur][aM][aK];
    v8bf xhi = *(const v8bf*)&xs[cur][aM][aK + 8];
    v8bf h0l = *(const v8bf*)&hb[cur][aM][aK];
    v8bf h0h = *(const v8bf*)&hb[cur][aM][aK + 8];
    v8bf h1l = *(const v8bf*)&hb[cur][aM][32 + aK];
    v8bf h1h = *(const v8bf*)&hb[cur][aM][32 + aK + 8];
    v16bf aX  = __builtin_shufflevector(xlo, xhi, 0,1,2,3,4,5,6,7,8,9,10,11,12,13,14,15);
    v16bf aH0 = __builtin_shufflevector(h0l, h0h, 0,1,2,3,4,5,6,7,8,9,10,11,12,13,14,15);
    v16bf aH1 = __builtin_shufflevector(h1l, h1h, 0,1,2,3,4,5,6,7,8,9,10,11,12,13,14,15);

    const int tp  = (t + 1 < TLEN) ? (t + 1) : (TLEN - 1);
    const int tok = tks[gr][tp];
    const float4 pf = ((const float4*)(emb + (size_t)tok * EMB))[gch];

    v8f acc[4];
#pragma unroll
    for (int g = 0; g < 4; ++g) {
      v8f a = {bgate[g], bgate[g], bgate[g], bgate[g], bgate[g], bgate[g], bgate[g], bgate[g]};
      a = __builtin_amdgcn_wmma_f32_16x16x32_bf16(false, aX,  false, bWk[g],  (short)0, a, false, false);
      a = __builtin_amdgcn_wmma_f32_16x16x32_bf16(false, aH0, false, bWr0[g], (short)0, a, false, false);
      a = __builtin_amdgcn_wmma_f32_16x16x32_bf16(false, aH1, false, bWr1[g], (short)0, a, false, false);
      acc[g] = a;
    }

#pragma unroll
    for (int r = 0; r < 8; ++r) {
      const float h = gate_update(acc[0][r], acc[1][r], acc[2][r], acc[3][r], creg[r]);
      hreg[r] = h;
      hb[nxt][mr + r][j] = (__bf16)h;
    }

    {
      v4bf p = {(__bf16)pf.x, (__bf16)pf.y, (__bf16)pf.z, (__bf16)pf.w};
      *(v4bf*)&xs[nxt][gr][gch * 4] = p;
    }

    __syncthreads();
  }

#pragma unroll
  for (int r = 0; r < 8; ++r) hf[mr + r][j] = hreg[r];
  __syncthreads();

  if (tid < MT * NCLS) {
    const int m = tid / NCLS, k = tid % NCLS;
    float acc = bd[k];
#pragma unroll
    for (int jj = 0; jj < HID; ++jj) acc += hf[m][jj] * Wd[jj * NCLS + k];
    lg[m][k] = acc;
  }
  __syncthreads();
  if (tid < MT) {
    const float l0 = lg[tid][0], l1 = lg[tid][1], l2 = lg[tid][2];
    const float mx = fmaxf(l0, fmaxf(l1, l2));
    const float e0 = __builtin_amdgcn_exp2f(LOG2E * (l0 - mx));
    const float e1 = __builtin_amdgcn_exp2f(LOG2E * (l1 - mx));
    const float e2 = __builtin_amdgcn_exp2f(LOG2E * (l2 - mx));
    const float inv = 1.f / (e0 + e1 + e2);
    float* o = out + (size_t)(b0 + tid) * NCLS;
    o[0] = e0 * inv; o[1] = e1 * inv; o[2] = e2 * inv;
  }
}

extern "C" void kernel_launch(void* const* d_in, const int* in_sizes, int n_in,
                              void* d_out, int out_size, void* d_ws, size_t ws_size,
                              hipStream_t stream) {
  const int*   tokens = (const int*)d_in[0];
  const float* emb    = (const float*)d_in[1];
  const float* Wk     = (const float*)d_in[2];
  const float* Wr     = (const float*)d_in[3];
  const float* b      = (const float*)d_in[4];
  const float* Wd     = (const float*)d_in[5];
  const float* bd     = (const float*)d_in[6];
  float* out = (float*)d_out;

  const size_t need = (size_t)VOCAB * NH4 * sizeof(float);   // 51.2 MB, L2-resident
  if (ws_size >= need) {
    float* zx = (float*)d_ws;
    emb_proj<<<VOCAB / MT, NTHR, 0, stream>>>(emb, Wk, b, zx);
    lstm_rec<<<BSZ / MT, NTHR, 0, stream>>>(tokens, zx, Wr, Wd, bd, out);
  } else {
    lstm_fused<<<BSZ / MT, NTHR, 0, stream>>>(tokens, emb, Wk, Wr, b, Wd, bd, out);
  }
}
